// MultiHeadSelfAttention_73143293051031
// MI455X (gfx1250) — matrix operand order repaired and checked by reading, not compiled, not run on hardware
//
#include <hip/hip_runtime.h>

#ifndef __has_builtin
#define __has_builtin(x) 0
#endif

typedef __attribute__((ext_vector_type(16))) __bf16       v16bf;
typedef __attribute__((ext_vector_type(8)))  float        v8f;
typedef __attribute__((ext_vector_type(4)))  unsigned int u32x4;
typedef __attribute__((ext_vector_type(4)))  float        f32x4;
typedef __attribute__((ext_vector_type(8)))  int          i32x8;
typedef __attribute__((ext_vector_type(4)))  int          i32x4;

#define BATCH 4
#define SEQ   2048
#define CDIM  1024
#define HEADS 16
#define HDIM  64
#define C3    3072
#define KCHUNK 64

#if __has_builtin(__builtin_amdgcn_tensor_load_to_lds) && \
    __has_builtin(__builtin_amdgcn_s_wait_tensorcnt)
#define USE_TDM 1
#else
#define USE_TDM 0
#endif

__device__ __forceinline__ unsigned short f32_to_bf16(float f) {
  unsigned int u = __float_as_uint(f);
  u += 0x7fffu + ((u >> 16) & 1u);  // round-to-nearest-even
  return (unsigned short)(u >> 16);
}
__device__ __forceinline__ unsigned int pack_bf16x2(float a, float b) {
  return (unsigned int)f32_to_bf16(a) | ((unsigned int)f32_to_bf16(b) << 16);
}
__device__ __forceinline__ v8f v8f_zero() {
  v8f v;
#pragma unroll
  for (int i = 0; i < 8; ++i) v[i] = 0.f;
  return v;
}
__device__ __forceinline__ float fast_exp2(float x) {
#if __has_builtin(__builtin_amdgcn_exp2f)
  return __builtin_amdgcn_exp2f(x);
#else
  return exp2f(x);
#endif
}

// 16-lane reductions via DPP16 (quad_perm xor1/xor2, half-mirror, mirror).
#if __has_builtin(__builtin_amdgcn_update_dpp)
#define HAVE_DPP 1
template <int CTRL>
__device__ __forceinline__ float dpp_permf(float x) {
  const int xi = __float_as_int(x);
  return __int_as_float(
      __builtin_amdgcn_update_dpp(xi, xi, CTRL, 0xF, 0xF, false));
}
#endif
__device__ __forceinline__ float red16_max(float x) {
#ifdef HAVE_DPP
  x = fmaxf(x, dpp_permf<0x0B1>(x));  // quad_perm [1,0,3,2]
  x = fmaxf(x, dpp_permf<0x04E>(x));  // quad_perm [2,3,0,1]
  x = fmaxf(x, dpp_permf<0x141>(x));  // row_half_mirror
  x = fmaxf(x, dpp_permf<0x140>(x));  // row_mirror
#else
#pragma unroll
  for (int off = 1; off < 16; off <<= 1) x = fmaxf(x, __shfl_xor(x, off, 32));
#endif
  return x;
}
__device__ __forceinline__ float red16_sum(float x) {
#ifdef HAVE_DPP
  x += dpp_permf<0x0B1>(x);
  x += dpp_permf<0x04E>(x);
  x += dpp_permf<0x141>(x);
  x += dpp_permf<0x140>(x);
#else
#pragma unroll
  for (int off = 1; off < 16; off <<= 1) x += __shfl_xor(x, off, 32);
#endif
  return x;
}

union Frag16 { u32x4 u[2]; v16bf bf; };

// ---------------------------------------------------------------------------
// GEMM: Out[m, n] = sum_k A[m, k] * W[n, k] + bias[n]
// A: [M x K] row-major, fp32 (AF32) or bf16; W: [Ncols x K] fp32 row-major.
// Out: bf16 (workspace) or fp32 (final), row-major [M x Ncols].
// Block: 256 threads = 8 waves; 128x64 output tile; wave = 32x32 (2x2 WMMA).
// ---------------------------------------------------------------------------
template <bool AF32, bool F32OUT>
__global__ __launch_bounds__(256) void gemm_bias_kernel(
    const void* __restrict__ Aptr, const float* __restrict__ W,
    const float* __restrict__ bias, void* __restrict__ Out,
    int M, int Ncols, int K) {
  __shared__ __align__(16) unsigned short As[128 * 32];
  __shared__ __align__(16) unsigned short Bs[64 * 32];

  const int t    = threadIdx.x;
  const int lane = t & 31;
  const int wid  = t >> 5;
  const int l16  = lane & 15;
  const bool hi  = lane >= 16;
  const int wm   = (wid & 3) * 32;
  const int wn   = (wid >> 2) * 32;
  const int bm   = blockIdx.y * 128;
  const int bn   = blockIdx.x * 64;

  v8f acc[2][2];
#pragma unroll
  for (int i = 0; i < 2; ++i)
#pragma unroll
    for (int j = 0; j < 2; ++j) acc[i][j] = v8f_zero();

  const int arow = t >> 1, acb = (t & 1) * 16;  // A tile: 128 rows x 32 k
  const int brow = t >> 2, bcb = (t & 3) * 8;   // B tile:  64 rows x 32 k

  for (int kk = 0; kk < K; kk += 32) {
    if (AF32) {
      const float* src = (const float*)Aptr + (size_t)(bm + arow) * K + kk + acb;
      f32x4 f0 = *(const f32x4*)(src + 0);
      f32x4 f1 = *(const f32x4*)(src + 4);
      f32x4 f2 = *(const f32x4*)(src + 8);
      f32x4 f3 = *(const f32x4*)(src + 12);
      u32x4 p0, p1;
      p0[0] = pack_bf16x2(f0[0], f0[1]); p0[1] = pack_bf16x2(f0[2], f0[3]);
      p0[2] = pack_bf16x2(f1[0], f1[1]); p0[3] = pack_bf16x2(f1[2], f1[3]);
      p1[0] = pack_bf16x2(f2[0], f2[1]); p1[1] = pack_bf16x2(f2[2], f2[3]);
      p1[2] = pack_bf16x2(f3[0], f3[1]); p1[3] = pack_bf16x2(f3[2], f3[3]);
      *(u32x4*)&As[arow * 32 + acb]     = p0;
      *(u32x4*)&As[arow * 32 + acb + 8] = p1;
    } else {
      const unsigned short* src =
          (const unsigned short*)Aptr + (size_t)(bm + arow) * K + kk + acb;
      *(u32x4*)&As[arow * 32 + acb]     = *(const u32x4*)(src);
      *(u32x4*)&As[arow * 32 + acb + 8] = *(const u32x4*)(src + 8);
    }
    {
      const float* src = W + (size_t)(bn + brow) * K + kk + bcb;
      f32x4 f0 = *(const f32x4*)(src + 0);
      f32x4 f1 = *(const f32x4*)(src + 4);
      u32x4 p;
      p[0] = pack_bf16x2(f0[0], f0[1]); p[1] = pack_bf16x2(f0[2], f0[3]);
      p[2] = pack_bf16x2(f1[0], f1[1]); p[3] = pack_bf16x2(f1[2], f1[3]);
      *(u32x4*)&Bs[brow * 32 + bcb] = p;
    }
    __syncthreads();

    Frag16 af[2], bf_[2];
#pragma unroll
    for (int mi = 0; mi < 2; ++mi) {
      const int row = wm + mi * 16 + l16;
      const int kb  = hi ? 8 : 0;
      af[mi].u[0] = *(const u32x4*)&As[row * 32 + kb];
      af[mi].u[1] = *(const u32x4*)&As[row * 32 + kb + 16];
    }
#pragma unroll
    for (int ni = 0; ni < 2; ++ni) {
      const int col = wn + ni * 16 + l16;
      const int kb  = hi ? 8 : 0;
      bf_[ni].u[0] = *(const u32x4*)&Bs[col * 32 + kb];
      bf_[ni].u[1] = *(const u32x4*)&Bs[col * 32 + kb + 16];
    }
#pragma unroll
    for (int mi = 0; mi < 2; ++mi)
#pragma unroll
      for (int ni = 0; ni < 2; ++ni)
        acc[mi][ni] = __builtin_amdgcn_wmma_f32_16x16x32_bf16(
            false, af[mi].bf, false, bf_[ni].bf, (short)0, acc[mi][ni], false,
            false);
    __syncthreads();
  }

#pragma unroll
  for (int mi = 0; mi < 2; ++mi)
#pragma unroll
    for (int ni = 0; ni < 2; ++ni) {
      const int col  = bn + wn + ni * 16 + l16;
      const float bv = bias[col];
#pragma unroll
      for (int r = 0; r < 8; ++r) {
        const int row = bm + wm + mi * 16 + r + (hi ? 8 : 0);
        const float v = acc[mi][ni][r] + bv;
        if (F32OUT)
          ((float*)Out)[(size_t)row * Ncols + col] = v;
        else
          ((unsigned short*)Out)[(size_t)row * Ncols + col] = f32_to_bf16(v);
      }
    }
}

// ---------------------------------------------------------------------------
// Flash attention: one block per (b, h, 128-query tile); wave = 16 queries.
// 64-key chunks; K staged via Tensor Data Mover, V via transpose copy.
// qkv: [B, SEQ, 3C] bf16; attn_out: [B, SEQ, C] bf16.
// ---------------------------------------------------------------------------
__global__ __launch_bounds__(256) void attn_kernel(
    const unsigned short* __restrict__ qkv,
    unsigned short* __restrict__ attn_out, float scale) {
  __shared__ __align__(16) unsigned short Kt[KCHUNK * 64];     // [key][d]
  __shared__ __align__(16) unsigned short Vt[64 * KCHUNK];     // [d][key]
  __shared__ __align__(16) unsigned short Pt[8][16 * KCHUNK];  // P per wave

  const int t    = threadIdx.x;
  const int lane = t & 31;
  const int wid  = t >> 5;
  const int l16  = lane & 15;
  const bool hi  = lane >= 16;
  const int bh   = blockIdx.y;
  const int b    = bh >> 4;
  const int h    = bh & 15;
  const int qbase = blockIdx.x * 128 + wid * 16;
  const float scl2 = scale * 1.4426950408889634f;  // fold log2(e)

  // Q fragments (A layout), d-slices 0-31 / 32-63, kept in registers.
  Frag16 qa[2];
  {
    const unsigned short* qp =
        qkv + ((size_t)(b * SEQ + qbase + l16)) * C3 + h * HDIM;
    const int kb = hi ? 8 : 0;
#pragma unroll
    for (int ks = 0; ks < 2; ++ks) {
      qa[ks].u[0] = *(const u32x4*)(qp + ks * 32 + kb);
      qa[ks].u[1] = *(const u32x4*)(qp + ks * 32 + kb + 16);
    }
  }

  v8f o[4];
#pragma unroll
  for (int i = 0; i < 4; ++i) o[i] = v8f_zero();
  float mrun[8], lrun[8];
#pragma unroll
  for (int r = 0; r < 8; ++r) { mrun[r] = -1e30f; lrun[r] = 0.f; }

  // V staging: paired keys so the transpose uses b32 DS stores.
  const int vkey = (t & 31) * 2;   // 0..62 (even)
  const int vdb  = (t >> 5) * 8;   // 0..56

#if USE_TDM
  const unsigned lds_kt = (unsigned)(unsigned long long)(&Kt[0]);
  const unsigned long long kbase0 = (unsigned long long)(
      qkv + ((size_t)b * SEQ) * C3 + CDIM + h * HDIM);
#endif

  for (int kc = 0; kc < SEQ; kc += KCHUNK) {
    // ---- K chunk: Tensor Data Mover (wave 0 issues the DMA) ----
#if USE_TDM
    if (wid == 0) {
      const unsigned long long ga =
          kbase0 + (unsigned long long)kc * (C3 * 2);
      u32x4 g0;
      g0[0] = 1u;                                   // count=1 valid descriptor
      g0[1] = lds_kt;                               // lds_addr
      g0[2] = (unsigned)ga;                         // global_addr[31:0]
      g0[3] = (unsigned)((ga >> 32) & 0x1ffffffu)   // global_addr[56:32]
              | (2u << 30);                         // type=2 (image)
      i32x8 g1;
      g1[0] = (int)(1u << 16);       // data_size=1 (2 bytes); mask=0; flags=0
      g1[1] = (int)(64u << 16);      // tensor_dim0[15:0] = 64 (hi half)
      g1[2] = (int)(64u << 16);      // tensor_dim0[31:16]=0 | tensor_dim1 lo=64
      g1[3] = (int)(64u << 16);      // tensor_dim1 hi=0 | tile_dim0=64
      g1[4] = 64;                    // tile_dim1=64 | tile_dim2=0
      g1[5] = C3;                    // tensor_dim0_stride[31:0] = 3072 elems
      g1[6] = 0;                     // stride0 hi | stride1 lo
      g1[7] = 0;                     // stride1 hi
      i32x4 z4;
      z4[0] = 0; z4[1] = 0; z4[2] = 0; z4[3] = 0;
#if defined(__clang_major__) && __clang_major__ >= 23
      i32x8 z8;
#pragma unroll
      for (int i = 0; i < 8; ++i) z8[i] = 0;
      __builtin_amdgcn_tensor_load_to_lds(g0, g1, z4, z4, z8, 0);
#else
      __builtin_amdgcn_tensor_load_to_lds(g0, g1, z4, z4, 0);
#endif
      __builtin_amdgcn_s_wait_tensorcnt(0);
    }
#else
    {
      const int kkey = t >> 2;          // 0..63
      const int kdb  = (t & 3) * 16;    // 0,16,32,48
      const unsigned short* kp = qkv +
          ((size_t)(b * SEQ + kc + kkey)) * C3 + CDIM + h * HDIM + kdb;
      *(u32x4*)&Kt[kkey * 64 + kdb]     = *(const u32x4*)kp;
      *(u32x4*)&Kt[kkey * 64 + kdb + 8] = *(const u32x4*)(kp + 8);
    }
#endif
    // ---- V chunk: cooperative transpose into Vt[d][key] ----
    {
      const unsigned short* vp = qkv +
          ((size_t)(b * SEQ + kc + vkey)) * C3 + 2 * CDIM + h * HDIM + vdb;
      u32x4 r0 = *(const u32x4*)vp;          // key vkey,   d vdb..vdb+7
      u32x4 r1 = *(const u32x4*)(vp + C3);   // key vkey+1, d vdb..vdb+7
#pragma unroll
      for (int j = 0; j < 8; ++j) {
        const unsigned e0 = (j & 1) ? (r0[j >> 1] >> 16) : (r0[j >> 1] & 0xffffu);
        const unsigned e1 = (j & 1) ? (r1[j >> 1] >> 16) : (r1[j >> 1] & 0xffffu);
        *(unsigned*)&Vt[(vdb + j) * KCHUNK + vkey] = e0 | (e1 << 16);
      }
    }
    __syncthreads();

    // ---- S = Q K^T: 4 key groups x 2 d-slices = 8 WMMA ----
    v8f s[4];
#pragma unroll
    for (int g = 0; g < 4; ++g) {
      s[g] = v8f_zero();
      const int key = g * 16 + l16;
      const int dof = hi ? 8 : 0;
#pragma unroll
      for (int ks = 0; ks < 2; ++ks) {
        Frag16 kf;
        kf.u[0] = *(const u32x4*)&Kt[key * 64 + ks * 32 + dof];
        kf.u[1] = *(const u32x4*)&Kt[key * 64 + ks * 32 + dof + 16];
        s[g] = __builtin_amdgcn_wmma_f32_16x16x32_bf16(
            false, qa[ks].bf, false, kf.bf, (short)0, s[g], false, false);
      }
    }

    // ---- online softmax (DPP 16-lane reductions, v_exp_f32 direct) ----
    unsigned short* Pw = Pt[wid];
#pragma unroll
    for (int r = 0; r < 8; ++r) {
      const float v0 = s[0][r] * scl2;
      const float v1 = s[1][r] * scl2;
      const float v2 = s[2][r] * scl2;
      const float v3 = s[3][r] * scl2;
      float mx = fmaxf(fmaxf(v0, v1), fmaxf(v2, v3));
      mx = red16_max(mx);
      const float mnew  = fmaxf(mrun[r], mx);
      const float alpha = fast_exp2(mrun[r] - mnew);
      mrun[r] = mnew;
      const float p0 = fast_exp2(v0 - mnew);
      const float p1 = fast_exp2(v1 - mnew);
      const float p2 = fast_exp2(v2 - mnew);
      const float p3 = fast_exp2(v3 - mnew);
      float rs = (p0 + p1) + (p2 + p3);
      rs = red16_sum(rs);
      lrun[r] = lrun[r] * alpha + rs;
#pragma unroll
      for (int i = 0; i < 4; ++i) o[i][r] *= alpha;
      const int prow = (r + (hi ? 8 : 0)) * KCHUNK + l16;
      Pw[prow +  0] = f32_to_bf16(p0);
      Pw[prow + 16] = f32_to_bf16(p1);
      Pw[prow + 32] = f32_to_bf16(p2);
      Pw[prow + 48] = f32_to_bf16(p3);
    }

    // ---- remap P (C layout -> A layout) via per-wave LDS tile ----
    Frag16 pf[2];
#pragma unroll
    for (int kc2 = 0; kc2 < 2; ++kc2) {
      const int base = l16 * KCHUNK + kc2 * 32 + (hi ? 8 : 0);
      pf[kc2].u[0] = *(const u32x4*)&Pw[base];
      pf[kc2].u[1] = *(const u32x4*)&Pw[base + 16];
    }

    // ---- O += P V: 4 d-groups x 2 key-slices = 8 WMMA ----
#pragma unroll
    for (int dg = 0; dg < 4; ++dg) {
      const int vrow = (dg * 16 + l16) * KCHUNK;
      const int kof  = hi ? 8 : 0;
#pragma unroll
      for (int kc2 = 0; kc2 < 2; ++kc2) {
        Frag16 vf;
        vf.u[0] = *(const u32x4*)&Vt[vrow + kc2 * 32 + kof];
        vf.u[1] = *(const u32x4*)&Vt[vrow + kc2 * 32 + kof + 16];
        o[dg] = __builtin_amdgcn_wmma_f32_16x16x32_bf16(
            false, pf[kc2].bf, false, vf.bf, (short)0, o[dg], false, false);
      }
    }
    __syncthreads();
  }

  // ---- normalize and store bf16 ----
#pragma unroll
  for (int dg = 0; dg < 4; ++dg) {
    const int d = dg * 16 + l16;
#pragma unroll
    for (int r = 0; r < 8; ++r) {
      const int q   = qbase + r + (hi ? 8 : 0);
      const float v = o[dg][r] / lrun[r];
      attn_out[((size_t)(b * SEQ + q)) * CDIM + h * HDIM + d] = f32_to_bf16(v);
    }
  }
}

// ---------------------------------------------------------------------------
extern "C" void kernel_launch(void* const* d_in, const int* in_sizes, int n_in,
                              void* d_out, int out_size, void* d_ws,
                              size_t ws_size, hipStream_t stream) {
  (void)in_sizes; (void)n_in; (void)out_size; (void)ws_size;
  const float* x      = (const float*)d_in[0];
  const float* qkv_w  = (const float*)d_in[1];
  const float* qkv_b  = (const float*)d_in[2];
  const float* proj_w = (const float*)d_in[3];
  const float* proj_b = (const float*)d_in[4];
  float* out = (float*)d_out;

  unsigned short* qkv_ws  = (unsigned short*)d_ws;                 // 48 MB
  unsigned short* attn_ws = qkv_ws + (size_t)BATCH * SEQ * C3;     // 16 MB

  const int M = BATCH * SEQ;  // 8192
  dim3 blk(256);

  // 1) QKV projection: x(f32) @ qkv_w^T + qkv_b -> bf16 [B, SEQ, 3C]
  gemm_bias_kernel<true, false><<<dim3(C3 / 64, M / 128), blk, 0, stream>>>(
      x, qkv_w, qkv_b, qkv_ws, M, C3, CDIM);

  // 2) Attention: flash-style per (b, h), scale = 1/sqrt(64)
  attn_kernel<<<dim3(SEQ / 128, BATCH * HEADS), blk, 0, stream>>>(
      qkv_ws, attn_ws, 0.125f);

  // 3) Output projection: attn(bf16) @ proj_w^T + proj_b -> f32 d_out
  gemm_bias_kernel<false, true><<<dim3(CDIM / 64, M / 128), blk, 0, stream>>>(
      attn_ws, proj_w, proj_b, out, M, CDIM, CDIM);
}
